// MyModel_31078383354745
// MI455X (gfx1250) — compile-verified, matrix operand order checked by reading, not run on hardware
//
#include <hip/hip_runtime.h>
#include <hip/hip_bf16.h>

// ---------------------------------------------------------------------------
// Types
// ---------------------------------------------------------------------------
typedef __bf16 bhalf;
typedef __bf16 v16bf __attribute__((ext_vector_type(16)));
typedef float  v8f   __attribute__((ext_vector_type(8)));
typedef int    v4i   __attribute__((vector_size(16)));   // matches async-LDS builtin

struct __align__(16) U16B { unsigned int x[4]; };          // POD 16-byte vector
union Frag16 { U16B q[2]; v16bf v; bhalf h[16]; };         // one WMMA A/B fragment
union Pack8  { U16B u;    bhalf h[8]; };

#define AS_GLOBAL __attribute__((address_space(1)))
#define AS_LDS    __attribute__((address_space(3)))

#if defined(__AMDGCN__) && __has_builtin(__builtin_amdgcn_global_load_async_to_lds_b128)
#define HAVE_ASYNC_LDS 1
#else
#define HAVE_ASYNC_LDS 0
#endif

__device__ __forceinline__ void wait_asynccnt0()
{
#if defined(__AMDGCN__)
#if __has_builtin(__builtin_amdgcn_s_wait_asynccnt)
  __builtin_amdgcn_s_wait_asynccnt(0);
#else
  asm volatile("s_wait_asynccnt 0" ::: "memory");
#endif
#endif
}

// Problem constants
constexpr int T   = 2048;
constexpr int IN  = 512;
constexpr int H   = 4096;
constexpr int O   = 1024;
constexpr int C   = 512;
constexpr int KIH = IN + H;   // 4608
constexpr int CH  = 16;       // time-chunk size
constexpr int NCH = T / CH;   // 128 chunks (with 1 zero-pad step)

// GEMM tiling
constexpr int BM = 128, BN = 128, BK = 32;

// ---------------------------------------------------------------------------
// Per-k-tile compute: fragment loads (CDNA5 16-bit A/B VGPR layout) + 8 WMMAs.
// Asb: [row][k] 128x32, Bsb: [col][k] 128x32 (i.e. B transposed).
// ---------------------------------------------------------------------------
__device__ __forceinline__ void mma_tile(const bhalf* __restrict__ Asb,
                                         const bhalf* __restrict__ Bsb,
                                         int wm, int wn, int laneM, int hiSel,
                                         v8f (&acc)[2][4])
{
  Frag16 af[2], bf[4];
#pragma unroll
  for (int i = 0; i < 2; ++i) {
    const U16B* p = (const U16B*)&Asb[(wm + i * 16 + laneM) * BK];
    af[i].q[0] = p[hiSel];
    af[i].q[1] = p[2 + hiSel];
  }
#pragma unroll
  for (int j = 0; j < 4; ++j) {
    const U16B* p = (const U16B*)&Bsb[(wn + j * 16 + laneM) * BK];
    bf[j].q[0] = p[hiSel];
    bf[j].q[1] = p[2 + hiSel];
  }
#pragma unroll
  for (int i = 0; i < 2; ++i)
#pragma unroll
    for (int j = 0; j < 4; ++j)
      acc[i][j] = __builtin_amdgcn_wmma_f32_16x16x32_bf16(
          false, af[i].v, false, bf[j].v, (short)0, acc[i][j], false, false);
}

// ---------------------------------------------------------------------------
// Generic bf16 GEMM with WMMA:  C[M,N] = A[M,K] * Bt[N,K]^T (+bias[m]) (+D[m,n])
//   A row-major (lda); Bt is B TRANSPOSED, row-major (ldbt) -> K contiguous.
//   Outputs: Cout (row-major, ldc) and/or CoutT (transposed, ldct), f32/bf16.
//   Requirements: M % 128 == 0, K % 32 == 0, Bt readable up to ceil(N/128)*128
//   rows (all callers satisfy this).
// Tiles are staged with GLOBAL_LOAD_ASYNC_TO_LDS_B128 (double-buffered,
// ASYNCcnt) when available; synchronous b128 copies otherwise.
// ---------------------------------------------------------------------------
__global__ __launch_bounds__(256)
void gemm_bf16_wmma(const bhalf* __restrict__ A, long lda,
                    const bhalf* __restrict__ Bt, long ldbt,
                    void* __restrict__ Cout, long ldc,
                    void* __restrict__ CoutT, long ldct,
                    int c_bf16,
                    const bhalf* __restrict__ Dadd, long lddm, long lddn,
                    const float* __restrict__ bias,
                    int N, int K)
{
  __shared__ __align__(16) bhalf As[2][BM * BK];
  __shared__ __align__(16) bhalf Bs[2][BN * BK];

  const int bm   = blockIdx.y * BM;
  const int bn   = blockIdx.x * BN;
  const int tid  = threadIdx.x;
  const int lane = tid & 31;
  const int wave = tid >> 5;
  const int wm   = (wave & 3) * 32;   // wave row offset inside tile
  const int wn   = (wave >> 2) * 64;  // wave col offset inside tile
  const int laneM = lane & 15;
  const int hiSel = lane >> 4;        // 0: K 0-7/16-23, 1: K 8-15/24-31

  // staging map: each thread owns 16 contiguous halfs of one row in each tile
  const int sr = tid >> 1;
  const int sc = (tid & 1) * 16;
  const bhalf* gA = A  + (long)(bm + sr) * lda  + sc;
  const bhalf* gB = Bt + (long)(bn + sr) * ldbt + sc;

  v8f acc[2][4];
#pragma unroll
  for (int i = 0; i < 2; ++i)
#pragma unroll
    for (int j = 0; j < 4; ++j)
#pragma unroll
      for (int e = 0; e < 8; ++e) acc[i][j][e] = 0.0f;

#if HAVE_ASYNC_LDS
  auto stage = [&](int buf, int k0) {
    AS_GLOBAL v4i* ga = (AS_GLOBAL v4i*)(gA + k0);
    AS_GLOBAL v4i* gb = (AS_GLOBAL v4i*)(gB + k0);
    AS_LDS v4i* la = (AS_LDS v4i*)&As[buf][sr * BK + sc];
    AS_LDS v4i* lb = (AS_LDS v4i*)&Bs[buf][sr * BK + sc];
    __builtin_amdgcn_global_load_async_to_lds_b128(ga,     la,     0, 0);
    __builtin_amdgcn_global_load_async_to_lds_b128(ga + 1, la + 1, 0, 0);
    __builtin_amdgcn_global_load_async_to_lds_b128(gb,     lb,     0, 0);
    __builtin_amdgcn_global_load_async_to_lds_b128(gb + 1, lb + 1, 0, 0);
  };
  const int NT = K / BK;
  stage(0, 0);
  wait_asynccnt0();
  __syncthreads();
  for (int t = 0; t < NT; ++t) {
    if (t + 1 < NT) stage((t + 1) & 1, (t + 1) * BK);     // prefetch next tile
    mma_tile(As[t & 1], Bs[t & 1], wm, wn, laneM, hiSel, acc);
    wait_asynccnt0();                                      // next tile landed
    __syncthreads();                                       // readers done, publish
  }
#else
  for (int k0 = 0; k0 < K; k0 += BK) {
    {
      U16B* dA = (U16B*)&As[0][sr * BK + sc];
      const U16B* sA = (const U16B*)(gA + k0);
      dA[0] = sA[0]; dA[1] = sA[1];
      U16B* dB = (U16B*)&Bs[0][sr * BK + sc];
      const U16B* sB = (const U16B*)(gB + k0);
      dB[0] = sB[0]; dB[1] = sB[1];
    }
    __syncthreads();
    mma_tile(As[0], Bs[0], wm, wn, laneM, hiSel, acc);
    __syncthreads();
  }
#endif

  // ---- epilogue: bias + strided bf16 addend; normal and/or transposed store
#pragma unroll
  for (int i = 0; i < 2; ++i) {
#pragma unroll
    for (int j = 0; j < 4; ++j) {
#pragma unroll
      for (int r = 0; r < 8; ++r) {
        const int m = bm + wm + i * 16 + r + (hiSel ? 8 : 0);
        const int n = bn + wn + j * 16 + laneM;
        if (n < N) {
          float v = acc[i][j][r];
          if (bias) v += bias[m];
          if (Dadd) v += (float)Dadd[(long)m * lddm + (long)n * lddn];
          if (Cout) {
            if (c_bf16) ((bhalf*)Cout)[(long)m * ldc + n] = (bhalf)v;
            else        ((float*)Cout)[(long)m * ldc + n] = v;
          }
          if (CoutT) {
            if (c_bf16) ((bhalf*)CoutT)[(long)n * ldct + m] = (bhalf)v;
            else        ((float*)CoutT)[(long)n * ldct + m] = v;
          }
        }
      }
    }
  }
}

// ---------------------------------------------------------------------------
// GEMV: y[m] = sum_k A[m,k]*x[k] (+ add[m*add_stride]) (+ bias[m])
// One wave per row, b128 loads + prefetch, shuffle reduction. K % 256 == 0.
// ---------------------------------------------------------------------------
__global__ __launch_bounds__(256)
void gemv_bf16(const bhalf* __restrict__ A, long lda, int M, int K,
               const bhalf* __restrict__ x,
               const float* __restrict__ add, long add_stride,
               const float* __restrict__ bias,
               float* __restrict__ out_f32, bhalf* __restrict__ out_bf16)
{
  const int lane = threadIdx.x & 31;
  const int wave = threadIdx.x >> 5;
  const int row  = blockIdx.x * 8 + wave;
  if (row >= M) return;

  const bhalf* a = A + (long)row * lda;
  float sum = 0.0f;
  for (int j = 0; j < K; j += 256) {
    const int k = j + lane * 8;
    if (j + 256 < K) __builtin_prefetch((const void*)(a + k + 256), 0, 0);
    Pack8 av, xv;
    av.u = *(const U16B*)(a + k);
    xv.u = *(const U16B*)(x + k);
#pragma unroll
    for (int e = 0; e < 8; ++e) sum += (float)av.h[e] * (float)xv.h[e];
  }
#pragma unroll
  for (int off = 16; off > 0; off >>= 1) sum += __shfl_down(sum, off, 32);

  if (lane == 0) {
    if (bias) sum += bias[row];
    if (add)  sum += add[(long)row * add_stride];
    if (out_f32)  out_f32[row]  = sum;
    if (out_bf16) out_bf16[row] = (bhalf)sum;
  }
}

// ---------------------------------------------------------------------------
// Small utility kernels
// ---------------------------------------------------------------------------
__global__ void cvt_f32_bf16(bhalf* __restrict__ dst, const float* __restrict__ src,
                             long sld, int rows, int cols)
{
  long i = (long)blockIdx.x * 256 + threadIdx.x;
  long n = (long)rows * cols;
  if (i < n) {
    long r = i / cols, c = i - r * cols;
    dst[i] = (bhalf)src[r * sld + c];
  }
}

// dst[r*cols + c] = src[c*sstride + r]   (transpose + convert)
__global__ void cvtT_f32_bf16(bhalf* __restrict__ dst, const float* __restrict__ src,
                              long sstride, int rows, int cols)
{
  long i = (long)blockIdx.x * 256 + threadIdx.x;
  long n = (long)rows * cols;
  if (i < n) {
    long r = i / cols, c = i - r * cols;
    dst[i] = (bhalf)src[c * sstride + r];
  }
}

// dst[r*cols + c] = src[r*srld + c*scst]
__global__ void gather_bf16(bhalf* __restrict__ dst, const bhalf* __restrict__ src,
                            int rows, int cols, long srld, long scst)
{
  long i = (long)blockIdx.x * 256 + threadIdx.x;
  long n = (long)rows * cols;
  if (i < n) {
    long r = i / cols, c = i - r * cols;
    dst[i] = src[r * srld + c * scst];
  }
}

__global__ void fill_strided_bf16(bhalf* __restrict__ dst, long n, long stride)
{
  long i = (long)blockIdx.x * 256 + threadIdx.x;
  if (i < n) dst[i * stride] = (bhalf)0.0f;
}

__global__ void copy_bf16(bhalf* __restrict__ dst, const bhalf* __restrict__ src, long n)
{
  long i = (long)blockIdx.x * 256 + threadIdx.x;
  if (i < n) dst[i] = src[i];
}

__global__ __launch_bounds__(256)
void logsoftmax_kernel(float* __restrict__ out, const float* __restrict__ in, int n)
{
  __shared__ float red[256];
  const int tid = threadIdx.x;
  float m = -1e30f;
  for (int i = tid; i < n; i += 256) m = fmaxf(m, in[i]);
  red[tid] = m; __syncthreads();
  for (int s = 128; s > 0; s >>= 1) {
    if (tid < s) red[tid] = fmaxf(red[tid], red[tid + s]);
    __syncthreads();
  }
  const float mx = red[0]; __syncthreads();
  float sum = 0.0f;
  for (int i = tid; i < n; i += 256) sum += __expf(in[i] - mx);
  red[tid] = sum; __syncthreads();
  for (int s = 128; s > 0; s >>= 1) {
    if (tid < s) red[tid] += red[tid + s];
    __syncthreads();
  }
  const float lse = mx + __logf(red[0]);
  for (int i = tid; i < n; i += 256) out[i] = in[i] - lse;
}

// ---------------------------------------------------------------------------
// Host orchestration
// ---------------------------------------------------------------------------
extern "C" void kernel_launch(void* const* d_in, const int* in_sizes, int n_in,
                              void* d_out, int out_size, void* d_ws, size_t ws_size,
                              hipStream_t stream)
{
  (void)in_sizes; (void)n_in; (void)out_size; (void)ws_size;

  const float* X[2]   = {(const float*)d_in[0], (const float*)d_in[1]};
  const float* Wih[2] = {(const float*)d_in[2], (const float*)d_in[6]};
  const float* bih[2] = {(const float*)d_in[3], (const float*)d_in[7]};
  const float* Wio[2] = {(const float*)d_in[4], (const float*)d_in[8]};
  const float* bio[2] = {(const float*)d_in[5], (const float*)d_in[9]};
  const float* Wd = (const float*)d_in[10];
  const float* bd = (const float*)d_in[11];

  // ---- workspace layout (~210 MB, 256B aligned) ----
  char* ws = (char*)d_ws;
  size_t off = 0;
  auto alloc = [&](size_t bytes) -> void* {
    void* p = ws + off;
    off += (bytes + 255) & ~(size_t)255;
    return p;
  };
  bhalf* WihB = (bhalf*)alloc((size_t)H * KIH * 2);     // bf16 Wih (Wx|Wh)
  bhalf* WioB = (bhalf*)alloc((size_t)O * KIH * 2);
  bhalf* WdB  = (bhalf*)alloc((size_t)C * 2 * O * 2);
  bhalf* Xb   = (bhalf*)alloc((size_t)T * IN * 2);      // X in bf16 (== B^T!)
  bhalf* U    = (bhalf*)alloc((size_t)H * T * 2);       // u'_j columns, col0 = 0
  bhalf* Pa   = (bhalf*)alloc((size_t)H * H * 2);       // power ping
  bhalf* PaT  = (bhalf*)alloc((size_t)H * H * 2);       // power ping, transposed
  bhalf* Pb   = (bhalf*)alloc((size_t)H * H * 2);       // power pong (ends = Wh^16)
  bhalf* PbT  = (bhalf*)alloc((size_t)H * H * 2);       // pong transposed / Wh^T seed
  bhalf* VbA  = (bhalf*)alloc((size_t)NCH * H * 2);     // Horner ping  [NCH, H]
  bhalf* VbB  = (bhalf*)alloc((size_t)NCH * H * 2);     // Horner pong  [NCH, H]
  float* Sf   = (float*)alloc((size_t)NCH * H * 4);     // chunk sums, [NCH, H]
  bhalf* hA   = (bhalf*)alloc((size_t)H * 2);
  bhalf* hB   = (bhalf*)alloc((size_t)H * 2);
  bhalf* clast = (bhalf*)alloc((size_t)KIH * 2);        // [x_last, h]
  float* outf[2];
  outf[0] = (float*)alloc((size_t)O * 4);
  outf[1] = (float*)alloc((size_t)O * 4);
  bhalf* catb   = (bhalf*)alloc((size_t)2 * O * 2);
  float* logits = (float*)alloc((size_t)C * 4);

  auto blocks1d = [](long n) { return dim3((unsigned)((n + 255) / 256)); };

  auto gemm = [&](const bhalf* A, long lda, const bhalf* Bt, long ldbt,
                  void* Cn, long ldc, void* Ct, long ldct, int cbf16,
                  const bhalf* D, long lddm, long lddn, const float* bias,
                  int M, int N, int K) {
    dim3 grid((N + BN - 1) / BN, M / BM);
    gemm_bf16_wmma<<<grid, dim3(256), 0, stream>>>(A, lda, Bt, ldbt, Cn, ldc,
                                                   Ct, ldct, cbf16,
                                                   D, lddm, lddn, bias, N, K);
  };
  auto gemv = [&](const bhalf* A, long lda, int M, int K, const bhalf* x,
                  const float* add, long add_stride, const float* bias,
                  float* of, bhalf* ob) {
    gemv_bf16<<<dim3((M + 7) / 8), dim3(256), 0, stream>>>(A, lda, M, K, x,
                                                           add, add_stride, bias, of, ob);
  };
  auto cvt = [&](bhalf* dst, const float* src, long sld, int rows, int cols) {
    cvt_f32_bf16<<<blocks1d((long)rows * cols), dim3(256), 0, stream>>>(dst, src, sld, rows, cols);
  };

  const long ldWh = KIH;           // Wh lives inside WihB at column offset IN

  for (int r = 0; r < 2; ++r) {
    // 0) convert weights / inputs to bf16
    cvt(WihB, Wih[r], KIH, H, KIH);
    cvt(WioB, Wio[r], KIH, O, KIH);
    cvt(Xb,   X[r],   IN,  T, IN);                       // B^T for the U-GEMM

    // 1) U' = [0 | Wx*X^T + bih] : zero-pad column 0, GEMM fills cols 1..T-1
    fill_strided_bf16<<<blocks1d(H), dim3(256), 0, stream>>>(U, H, T);
    gemm(WihB, KIH, Xb, IN, (void*)(U + 1), T, nullptr, 0, 1,
         nullptr, 0, 0, bih[r], H, T - 1, IN);

    // 2) Wh^16 via 4 squarings; each GEMM but the last emits the power in both orientations
    const bhalf* Wh = WihB + IN;
    cvtT_f32_bf16<<<blocks1d((long)H * H), dim3(256), 0, stream>>>(PbT, Wih[r] + IN, KIH, H, H); // Wh^T
    gemm(Wh, ldWh, PbT, H, Pa, H, PaT, H, 1, nullptr, 0, 0, nullptr, H, H, H); // Wh^2
    gemm(Pa, H,    PaT, H, Pb, H, PbT, H, 1, nullptr, 0, 0, nullptr, H, H, H); // Wh^4
    gemm(Pb, H,    PbT, H, Pa, H, PaT, H, 1, nullptr, 0, 0, nullptr, H, H, H); // Wh^8
    gemm(Pa, H,    PaT, H, Pb, H, nullptr, 0, 1, nullptr, 0, 0, nullptr, H, H, H); // Wh^16

    // 3) chunk sums s_c = sum_i Wh^{15-i} u'_{16c+i} by Horner over all 128
    //    chunks at once: V <- Wh*V + U'_i (i = 1..15); V kept in B^T layout.
    gather_bf16<<<blocks1d((long)NCH * H), dim3(256), 0, stream>>>(VbA, U, NCH, H, 16, T);
    for (int i = 1; i <= 15; ++i) {
      const bhalf* src = (i & 1) ? VbA : VbB;
      if (i < 15) {
        bhalf* dst = (i & 1) ? VbB : VbA;
        gemm(Wh, ldWh, src, H, nullptr, 0, dst, H, 1, U + i, T, CH, nullptr, H, NCH, H);
      } else {
        gemm(Wh, ldWh, src, H, nullptr, 0, Sf, H, 0, U + i, T, CH, nullptr, H, NCH, H);
      }
    }

    // 4) sequential carry over chunks: h <- Wh^16 * h + s_c  (Wh^16 L2-resident)
    fill_strided_bf16<<<blocks1d(H), dim3(256), 0, stream>>>(hA, H, 1);
    for (int c = 0; c < NCH; ++c) {
      const bhalf* hs = (c & 1) ? hB : hA;
      bhalf*       hd = (c & 1) ? hA : hB;
      gemv(Pb, H, H, H, hs, Sf + (long)c * H, 1, nullptr, nullptr, hd);
    }
    // after c = 127 (odd), final h is in hA

    // 5) i2o: out = Wio * [x_last, h] + bio
    cvt(clast, X[r] + (size_t)(T - 1) * IN, IN, 1, IN);
    copy_bf16<<<blocks1d(H), dim3(256), 0, stream>>>(clast + IN, hA, H);
    gemv(WioB, KIH, O, KIH, clast, nullptr, 0, bio[r], outf[r], nullptr);
  }

  // 6) decoder + log_softmax
  cvt(WdB, Wd, 2 * O, C, 2 * O);
  cvt(catb,     outf[0], O, 1, O);
  cvt(catb + O, outf[1], O, 1, O);
  gemv(WdB, 2 * O, C, 2 * O, catb, nullptr, 0, bd, logits, nullptr);
  logsoftmax_kernel<<<dim3(1), dim3(256), 0, stream>>>((float*)d_out, logits, C);
}
